// Mamba2Mixer_43542378447400
// MI455X (gfx1250) — compile-verified, matrix operand order checked by reading, not run on hardware
//
#include <hip/hip_runtime.h>
#include <hip/hip_bf16.h>

// ---------------- problem constants ----------------
#define H_     128
#define P_     64
#define G_     8
#define N_     128
#define CS_    128
#define KCONV_ 4
#define DIM_   4096
#define L_     2048
#define CN_    16          // L / CS
#define INTER_ 8192        // H*P
#define CONVD_ 10240       // INTER + 2*G*N
#define NPROJ_ 18560       // INTER + CONVD + H
#define EPS_   1e-5f

typedef __bf16 bf16;
typedef __attribute__((ext_vector_type(16))) __bf16 v16bf;
typedef __attribute__((ext_vector_type(8)))  float  v8f;

union FragU { v16bf v; uint4 q[2]; };

__device__ __forceinline__ int lane_id() { return (int)(threadIdx.x & 31); }

// Load a 16x32 bf16 WMMA operand fragment.
// CDNA5 16-bit A/B layout: lane&15 = fixed (row/col) index, lane>>4 = k-group;
// elements 0..7 at k0+kb, 8..15 at k0+kb+16 (kb = (lane>>4)*8).
__device__ __forceinline__ v16bf frag_ld(const bf16* p0, long stride, int k0) {
  const int lane = lane_id();
  const bf16* r = p0 + (long)(lane & 15) * stride + k0 + ((lane >> 4) << 3);
  FragU f;
  f.q[0] = *(const uint4*)(r);
  f.q[1] = *(const uint4*)(r + 16);
  return f.v;
}

__device__ __forceinline__ v8f wmma_bf(v16bf a, v16bf b, v8f c) {
  return __builtin_amdgcn_wmma_f32_16x16x32_bf16(false, a, false, b, (short)0, c,
                                                 false, false);
}

__device__ __forceinline__ v8f zero8() {
  v8f z;
#pragma unroll
  for (int e = 0; e < 8; ++e) z[e] = 0.f;
  return z;
}

__device__ __forceinline__ float siluf(float x) { return x / (1.f + expf(-x)); }

// ---------------- f32 -> bf16 bulk convert (4 elems/thread) -----------------
__global__ __launch_bounds__(256) void cvt_bf16_kernel(const float* __restrict__ src,
                                                       bf16* __restrict__ dst) {
  long i = ((long)blockIdx.x * 256 + threadIdx.x) * 4;
  float4 v = *(const float4*)(src + i);
  dst[i + 0] = (bf16)v.x;
  dst[i + 1] = (bf16)v.y;
  dst[i + 2] = (bf16)v.z;
  dst[i + 3] = (bf16)v.w;
}

// ---------------- GEMM: C = A * B^T, A[M,K] bf16, B[N,K] bf16 ---------------
#define LDP   72           // LDS pitch (bf16): 144B rows, 16B aligned, no conflicts
#define TILEB (128 * LDP)  // one 128x64 staging buffer (bf16 elements)

// Pure-copy staging via CDNA5 async load-to-LDS (ASYNCcnt path).
// Each thread copies 64B: row = t>>1 (128 rows), c0 = (t&1)*32 of a 128x64 tile.
__device__ __forceinline__ void stage_async(bf16* dst_lds, const bf16* src, long ld) {
  const int t = threadIdx.x;
  const int row = t >> 1, c0 = (t & 1) << 5;
  const bf16* s = src + (long)row * ld + c0;
  unsigned d = (unsigned)(uintptr_t)(dst_lds + row * LDP + c0);
#pragma unroll
  for (int j = 0; j < 4; ++j) {
    asm volatile("global_load_async_to_lds_b128 %0, %1, off"
                 :: "v"(d + 16 * j), "v"(s + 8 * j) : "memory");
  }
}

__global__ __launch_bounds__(256) void gemm_bt(const bf16* __restrict__ A,
                                               const bf16* __restrict__ B,
                                               float* __restrict__ C,
                                               int K, long lda, long ldb, long ldc) {
  __shared__ bf16 As[2 * TILEB];
  __shared__ bf16 Bs[2 * TILEB];
  const long m0 = (long)blockIdx.y * 128;
  const long n0 = (long)blockIdx.x * 128;
  const int wave = threadIdx.x >> 5;
  const int wm = wave & 3;       // 4 waves over M: rows wm*32..+32
  const int wn = wave >> 2;      // 2 waves over N: cols wn*64..+64
  v8f acc[2][4];
#pragma unroll
  for (int i = 0; i < 2; ++i)
#pragma unroll
    for (int j = 0; j < 4; ++j) acc[i][j] = zero8();

  // prologue: stage first K-tile into buffer 0
  stage_async(As, A + m0 * lda, lda);
  stage_async(Bs, B + n0 * ldb, ldb);

  int cur = 0;
  for (int k0 = 0; k0 < K; k0 += 64) {
    // wait for the stage issued one compute-phase ago (async ops are in-order)
    asm volatile("s_wait_asynccnt 0" ::: "memory");
    __syncthreads();
    // kick off next tile's DMA into the other buffer; overlaps with WMMAs below
    if (k0 + 64 < K) {
      const int nxt = cur ^ 1;
      stage_async(As + nxt * TILEB, A + m0 * lda + k0 + 64, lda);
      stage_async(Bs + nxt * TILEB, B + n0 * ldb + k0 + 64, ldb);
    }
    const bf16* Ab = As + cur * TILEB;
    const bf16* Bb = Bs + cur * TILEB;
#pragma unroll
    for (int kk = 0; kk < 64; kk += 32) {
      v16bf af[2], bfm[4];
#pragma unroll
      for (int i = 0; i < 2; ++i) af[i]  = frag_ld(Ab + (wm * 32 + i * 16) * LDP, LDP, kk);
#pragma unroll
      for (int j = 0; j < 4; ++j) bfm[j] = frag_ld(Bb + (wn * 64 + j * 16) * LDP, LDP, kk);
#pragma unroll
      for (int i = 0; i < 2; ++i)
#pragma unroll
        for (int j = 0; j < 4; ++j) acc[i][j] = wmma_bf(af[i], bfm[j], acc[i][j]);
    }
    cur ^= 1;
  }

  const int lane = lane_id();
  const int col = lane & 15;
  const int rb  = (lane >> 4) << 3;   // C/D layout: row = r + 8*(lane>=16)
#pragma unroll
  for (int i = 0; i < 2; ++i)
#pragma unroll
    for (int j = 0; j < 4; ++j)
#pragma unroll
      for (int r = 0; r < 8; ++r) {
        long row = m0 + wm * 32 + i * 16 + r + rb;
        long cc  = n0 + wn * 64 + j * 16 + col;
        C[row * ldc + cc] = acc[i][j][r];
      }
}

// ---------------- causal depthwise conv (K=4) + SiLU ------------------------
__global__ void conv_silu_kernel(const float* __restrict__ proj,
                                 const float* __restrict__ cw,
                                 const float* __restrict__ cb,
                                 float* __restrict__ hbc) {
  long idx = (long)blockIdx.x * 256 + threadIdx.x;
  int ch = (int)(idx % CONVD_);
  long l = idx / CONVD_;
  float s = cb[ch];
#pragma unroll
  for (int k = 0; k < KCONV_; ++k) {
    long ls = l - (KCONV_ - 1) + k;
    if (ls >= 0) s += cw[ch * KCONV_ + k] * proj[ls * NPROJ_ + INTER_ + ch];
  }
  hbc[idx] = siluf(s);
}

// ---------------- dt = softplus(dt_raw + bias) ------------------------------
__global__ void dt_kernel(const float* __restrict__ proj,
                          const float* __restrict__ dtb,
                          float* __restrict__ dtsp) {
  long idx = (long)blockIdx.x * 256 + threadIdx.x;
  int h = (int)(idx % H_);
  long l = idx / H_;
  float v = proj[l * NPROJ_ + (INTER_ + CONVD_) + h] + dtb[h];
  dtsp[idx] = (v > 20.f) ? v : log1pf(expf(v));
}

// ---------------- per-(chunk, head) inclusive cumsum of A*dt ----------------
__global__ void acum_kernel(const float* __restrict__ dtsp,
                            const float* __restrict__ alog,
                            float* __restrict__ acum) {
  const int c = blockIdx.x / H_, h = blockIdx.x % H_;
  __shared__ float s[CS_];
  const int t = threadIdx.x;
  float A = -expf(alog[h]);
  s[t] = A * dtsp[((long)c * CS_ + t) * H_ + h];
  __syncthreads();
  for (int off = 1; off < CS_; off <<= 1) {
    float v = (t >= off) ? s[t - off] : 0.f;
    __syncthreads();
    s[t] += v;
    __syncthreads();
  }
  acum[(long)blockIdx.x * CS_ + t] = s[t];
}

// ---------------- x*dt and x*dt*decay, LDS tile transpose -> [h][p][l] ------
__global__ __launch_bounds__(256) void xdx_kernel(const float* __restrict__ hbc,
                                                  const float* __restrict__ dtsp,
                                                  const float* __restrict__ acum,
                                                  bf16* __restrict__ xdtT,
                                                  bf16* __restrict__ xdecT) {
  __shared__ float xt[64 * 65];
  __shared__ float fac[64];
  __shared__ float dts[64];
  const long l0 = (long)blockIdx.x * 64;
  const int h = blockIdx.y;
  const int t = threadIdx.x;
  if (t < 64) {
    long l = l0 + t;
    int c = (int)(l >> 7), lc = (int)(l & 127);
    const float* ac = acum + ((long)c * H_ + h) * CS_;
    fac[t] = expf(ac[CS_ - 1] - ac[lc]);
    dts[t] = dtsp[l * H_ + h];
  }
  __syncthreads();
#pragma unroll
  for (int it = 0; it < 16; ++it) {            // coalesced read: p fastest
    int o = it * 256 + t;
    int l = o >> 6, p = o & 63;
    xt[l * 65 + p] = hbc[(l0 + l) * CONVD_ + h * P_ + p] * dts[l];
  }
  __syncthreads();
#pragma unroll
  for (int it = 0; it < 16; ++it) {            // coalesced write: l fastest
    int o = it * 256 + t;
    int p = o >> 6, l = o & 63;
    float v = xt[l * 65 + p];
    long ofs = ((long)(h * P_ + p)) * L_ + l0 + l;
    xdtT[ofs]  = (bf16)v;
    xdecT[ofs] = (bf16)(v * fac[l]);
  }
}

// ---------------- bf16 copies of B, C (row-major) ---------------------------
__global__ void bc_kernel(const float* __restrict__ hbc,
                          bf16* __restrict__ Bb, bf16* __restrict__ Cb) {
  long idx = (long)blockIdx.x * 256 + threadIdx.x;
  int gn = (int)(idx % (G_ * N_));
  long l = idx / (G_ * N_);
  Bb[idx] = (bf16)hbc[l * CONVD_ + INTER_ + gn];
  Cb[idx] = (bf16)hbc[l * CONVD_ + INTER_ + G_ * N_ + gn];
}

// ---------------- B transposed -> [g*N+n][l] via LDS tiles ------------------
__global__ __launch_bounds__(256) void btrans_kernel(const float* __restrict__ hbc,
                                                     bf16* __restrict__ BT) {
  __shared__ float tb[64 * 65];
  const long l0 = (long)blockIdx.x * 64;
  const int g0 = blockIdx.y * 64;
  const int t = threadIdx.x;
#pragma unroll
  for (int it = 0; it < 16; ++it) {
    int o = it * 256 + t;
    int l = o >> 6, n = o & 63;
    tb[l * 65 + n] = hbc[(l0 + l) * CONVD_ + INTER_ + g0 + n];
  }
  __syncthreads();
#pragma unroll
  for (int it = 0; it < 16; ++it) {
    int o = it * 256 + t;
    int n = o >> 6, l = o & 63;
    BT[(long)(g0 + n) * L_ + l0 + l] = (bf16)tb[l * 65 + n];
  }
}

// ---------------- per-(c,h) chunk states = xdec^T . B (64x128, WMMA) --------
__global__ __launch_bounds__(256) void states_kernel(const bf16* __restrict__ xdecT,
                                                     const bf16* __restrict__ BT,
                                                     float* __restrict__ states) {
  const int c = blockIdx.x, h = blockIdx.y, g = h >> 4;
  const int wave = threadIdx.x >> 5, lane = threadIdx.x & 31;
  const int mt = wave & 3;        // p tile (4 of 16 -> P=64)
  const int nhalf = wave >> 2;    // 0..1 -> n tiles nhalf*4 + j
  const int col = lane & 15, rb = (lane >> 4) << 3;
  v8f acc[4];
#pragma unroll
  for (int j = 0; j < 4; ++j) acc[j] = zero8();
  const bf16* Abase = xdecT + ((long)(h * P_ + mt * 16)) * L_ + c * CS_;
  for (int k0 = 0; k0 < CS_; k0 += 32) {
    v16bf a = frag_ld(Abase, L_, k0);
#pragma unroll
    for (int j = 0; j < 4; ++j) {
      v16bf b = frag_ld(BT + ((long)(g * N_ + nhalf * 64 + j * 16)) * L_ + c * CS_, L_, k0);
      acc[j] = wmma_bf(a, b, acc[j]);
    }
  }
  float* S = states + ((long)c * H_ + h) * (P_ * N_);
#pragma unroll
  for (int j = 0; j < 4; ++j)
#pragma unroll
    for (int r = 0; r < 8; ++r) {
      int p = mt * 16 + r + rb;
      int n = nhalf * 64 + j * 16 + col;
      S[p * N_ + n] = acc[j][r];
    }
}

// ---------------- inter-chunk state recurrence (per element) ----------------
__global__ void scan_kernel(const float* __restrict__ states,
                            const float* __restrict__ acum,
                            bf16* __restrict__ prevb) {
  long e = (long)blockIdx.x * 256 + threadIdx.x;   // element in [H][P][N]
  int h = (int)(e >> 13);                          // P*N = 8192
  float run = 0.f;
  for (int c = 0; c < CN_; ++c) {
    long o = (long)c * ((long)H_ * P_ * N_) + e;
    prevb[o] = (bf16)run;                          // state entering chunk c
    run = run * expf(acum[((long)c * H_ + h) * CS_ + (CS_ - 1)]) + states[o];
  }
}

// ---------------- per-(c,h) Y = M.xdt + exp(Acum).(C.prev^T) + D.x ----------
__global__ __launch_bounds__(256) void ychunk_kernel(const bf16* __restrict__ Cb,
                                                     const bf16* __restrict__ Bb,
                                                     const bf16* __restrict__ prevb,
                                                     const bf16* __restrict__ xdtT,
                                                     const float* __restrict__ acum,
                                                     const float* __restrict__ hbc,
                                                     const float* __restrict__ Dp,
                                                     float* __restrict__ y) {
  __shared__ float acs[CS_];
  __shared__ bf16 Ml[CS_ * 136];               // 272B rows: 16B aligned, no conflicts
  const int c = blockIdx.x, h = blockIdx.y, g = h >> 4;
  const int tid = threadIdx.x, wave = tid >> 5, lane = tid & 31;
  const int col = lane & 15, rb = (lane >> 4) << 3;
  if (tid < CS_) acs[tid] = acum[((long)c * H_ + h) * CS_ + tid];
  __syncthreads();

  // Phase 1: G = C.B^T ; M = G .* exp(segsum) (lower-tri) -> LDS bf16
  {
    const int mt = wave;                        // 8 waves x 8 n-tiles
    const bf16* Abase = Cb + ((long)(c * CS_ + mt * 16)) * (G_ * N_) + g * N_;
    v8f gacc[8];
#pragma unroll
    for (int j = 0; j < 8; ++j) gacc[j] = zero8();
    for (int k0 = 0; k0 < N_; k0 += 32) {
      v16bf a = frag_ld(Abase, G_ * N_, k0);
#pragma unroll
      for (int nt = 0; nt < 8; ++nt) {
        v16bf b = frag_ld(Bb + ((long)(c * CS_ + nt * 16)) * (G_ * N_) + g * N_,
                          G_ * N_, k0);
        gacc[nt] = wmma_bf(a, b, gacc[nt]);
      }
    }
#pragma unroll
    for (int nt = 0; nt < 8; ++nt)
#pragma unroll
      for (int r = 0; r < 8; ++r) {
        int l = mt * 16 + r + rb;
        int s = nt * 16 + col;
        float v = (s <= l) ? gacc[nt][r] * expf(acs[l] - acs[s]) : 0.f;
        Ml[l * 136 + s] = (bf16)v;
      }
  }
  __syncthreads();

  // Phase 2: Y_off = C.prev^T (scaled by exp(Acum[l])), then Y_diag = M.xdt
  {
    const int mt = wave;
    v8f yacc[4];
#pragma unroll
    for (int j = 0; j < 4; ++j) yacc[j] = zero8();
    const bf16* Abase = Cb + ((long)(c * CS_ + mt * 16)) * (G_ * N_) + g * N_;
    const bf16* Pbase = prevb + ((long)c * H_ + h) * (P_ * N_);
    for (int k0 = 0; k0 < N_; k0 += 32) {
      v16bf a = frag_ld(Abase, G_ * N_, k0);
#pragma unroll
      for (int j = 0; j < 4; ++j) {
        v16bf b = frag_ld(Pbase + (long)(j * 16) * N_, N_, k0);
        yacc[j] = wmma_bf(a, b, yacc[j]);
      }
    }
#pragma unroll
    for (int j = 0; j < 4; ++j)
#pragma unroll
      for (int r = 0; r < 8; ++r) {
        int l = mt * 16 + r + rb;
        yacc[j][r] *= expf(acs[l]);
      }
    const bf16* Mbase = Ml + mt * 16 * 136;
    for (int k0 = 0; k0 < CS_; k0 += 32) {
      v16bf a = frag_ld(Mbase, 136, k0);
#pragma unroll
      for (int j = 0; j < 4; ++j) {
        v16bf b = frag_ld(xdtT + ((long)(h * P_ + j * 16)) * L_ + c * CS_, L_, k0);
        yacc[j] = wmma_bf(a, b, yacc[j]);
      }
    }
    float Dh = Dp[h];
#pragma unroll
    for (int j = 0; j < 4; ++j)
#pragma unroll
      for (int r = 0; r < 8; ++r) {
        int l = mt * 16 + r + rb;
        int p = j * 16 + col;
        long li = (long)c * CS_ + l;
        float xv = hbc[li * CONVD_ + h * P_ + p];
        y[li * INTER_ + h * P_ + p] = yacc[j][r] + Dh * xv;
      }
  }
}

// ---------------- RMSnorm * norm_w * SiLU(gate) -> bf16 ---------------------
__global__ __launch_bounds__(256) void rms_gate_kernel(const float* __restrict__ y,
                                                       const float* __restrict__ proj,
                                                       const float* __restrict__ nw,
                                                       bf16* __restrict__ yb) {
  __shared__ float red[256];
  const long l = blockIdx.x;
  const float* yr = y + l * INTER_;
  float ss = 0.f;
  for (int i = threadIdx.x; i < INTER_; i += 256) { float v = yr[i]; ss += v * v; }
  red[threadIdx.x] = ss;
  __syncthreads();
  for (int o = 128; o > 0; o >>= 1) {
    if ((int)threadIdx.x < o) red[threadIdx.x] += red[threadIdx.x + o];
    __syncthreads();
  }
  float rs = rsqrtf(red[0] / (float)INTER_ + EPS_);
  const float* gr = proj + l * NPROJ_;
  bf16* orow = yb + l * INTER_;
  for (int i = threadIdx.x; i < INTER_; i += 256) {
    float v = yr[i] * rs * nw[i];
    orow[i] = (bf16)(v * siluf(gr[i]));
  }
}

// ---------------- host: pipeline ----------------
extern "C" void kernel_launch(void* const* d_in, const int* in_sizes, int n_in,
                              void* d_out, int out_size, void* d_ws, size_t ws_size,
                              hipStream_t stream) {
  const float* hs   = (const float*)d_in[0];
  const float* win  = (const float*)d_in[1];
  const float* cw   = (const float*)d_in[2];
  const float* cb   = (const float*)d_in[3];
  const float* dtb  = (const float*)d_in[4];
  const float* alog = (const float*)d_in[5];
  const float* Dp   = (const float*)d_in[6];
  const float* nw   = (const float*)d_in[7];
  const float* wout = (const float*)d_in[8];
  float* out = (float*)d_out;

  char* w = (char*)d_ws;
  auto alloc = [&](size_t n) { void* p = (void*)w; w += (n + 255) & ~(size_t)255; return p; };
  float* proj   = (float*)alloc((size_t)L_ * NPROJ_ * 4);
  float* hbc    = (float*)alloc((size_t)L_ * CONVD_ * 4);
  float* dtsp   = (float*)alloc((size_t)L_ * H_ * 4);
  float* acum   = (float*)alloc((size_t)CN_ * H_ * CS_ * 4);
  bf16*  hsb    = (bf16*)alloc((size_t)L_ * DIM_ * 2);
  bf16*  winb   = (bf16*)alloc((size_t)NPROJ_ * DIM_ * 2);
  bf16*  woutb  = (bf16*)alloc((size_t)DIM_ * INTER_ * 2);
  bf16*  xdtT   = (bf16*)alloc((size_t)H_ * P_ * L_ * 2);
  bf16*  xdecT  = (bf16*)alloc((size_t)H_ * P_ * L_ * 2);
  bf16*  Bb     = (bf16*)alloc((size_t)L_ * G_ * N_ * 2);
  bf16*  Cb     = (bf16*)alloc((size_t)L_ * G_ * N_ * 2);
  bf16*  BT     = (bf16*)alloc((size_t)G_ * N_ * L_ * 2);
  float* states = (float*)alloc((size_t)CN_ * H_ * P_ * N_ * 4);
  bf16*  prevb  = (bf16*)alloc((size_t)CN_ * H_ * P_ * N_ * 2);
  float* ybuf   = (float*)alloc((size_t)L_ * INTER_ * 4);
  bf16*  ybf    = (bf16*)alloc((size_t)L_ * INTER_ * 2);

  // 0) one-shot bf16 conversions: streamed GEMM operands become L2-resident
  cvt_bf16_kernel<<<((long)L_ * DIM_) / 1024, 256, 0, stream>>>(hs, hsb);
  cvt_bf16_kernel<<<((long)NPROJ_ * DIM_) / 1024, 256, 0, stream>>>(win, winb);
  cvt_bf16_kernel<<<((long)DIM_ * INTER_) / 1024, 256, 0, stream>>>(wout, woutb);

  // 1) proj = hs @ Win^T   (2048 x 18560, K=4096)
  gemm_bt<<<dim3(NPROJ_ / 128, L_ / 128), 256, 0, stream>>>(
      hsb, winb, proj, DIM_, (long)DIM_, (long)DIM_, (long)NPROJ_);
  // 2) causal conv + silu
  conv_silu_kernel<<<((long)L_ * CONVD_) / 256, 256, 0, stream>>>(proj, cw, cb, hbc);
  // 3) dt softplus
  dt_kernel<<<(L_ * H_) / 256, 256, 0, stream>>>(proj, dtb, dtsp);
  // 4) A cumsum per (chunk, head)
  acum_kernel<<<CN_ * H_, CS_, 0, stream>>>(dtsp, alog, acum);
  // 5) xdt / xdecay transposed bf16 (LDS tile transpose)
  xdx_kernel<<<dim3(L_ / 64, H_), 256, 0, stream>>>(hbc, dtsp, acum, xdtT, xdecT);
  // 6) B/C bf16 + B transposed
  bc_kernel<<<((long)L_ * G_ * N_) / 256, 256, 0, stream>>>(hbc, Bb, Cb);
  btrans_kernel<<<dim3(L_ / 64, (G_ * N_) / 64), 256, 0, stream>>>(hbc, BT);
  // 7) chunk states (WMMA)
  states_kernel<<<dim3(CN_, H_), 256, 0, stream>>>(xdecT, BT, states);
  // 8) inter-chunk recurrence (element-parallel)
  scan_kernel<<<(H_ * P_ * N_) / 256, 256, 0, stream>>>(states, acum, prevb);
  // 9) Y_diag + Y_off + D residual (WMMA)
  ychunk_kernel<<<dim3(CN_, H_), 256, 0, stream>>>(Cb, Bb, prevb, xdtT, acum, hbc, Dp, ybuf);
  // 10) RMSnorm + gate
  rms_gate_kernel<<<L_, 256, 0, stream>>>(ybuf, proj, nw, ybf);
  // 11) out = y @ Wout^T   (2048 x 4096, K=8192)
  gemm_bt<<<dim3(DIM_ / 128, L_ / 128), 256, 0, stream>>>(
      ybf, woutb, out, INTER_, (long)INTER_, (long)INTER_, (long)DIM_);

  (void)in_sizes; (void)n_in; (void)out_size; (void)ws_size;
}
